// HanDecoderSingleZ_42743514530129
// MI455X (gfx1250) — matrix operand order repaired and checked by reading, not compiled, not run on hardware
//
#include <hip/hip_runtime.h>
#include <hip/hip_bf16.h>
#include <math.h>
#include <stdint.h>

// ---------------------------------------------------------------------------
// Problem constants (from the reference)
// ---------------------------------------------------------------------------
#define N_NOTES   2000
#define OUTF      10          // OUT - 1
#define ENC       64
#define ENCV      16
#define BEAT_SZ   128
#define FINAL     128
#define NOTE_D    256
#define BEAT_D    256
#define MEAS_D    128
#define RES_D     8

#define T_IN      467
#define F_IN      715

// GEMM tiling (wave32 WMMA 16x16x32 f16)
#define MT        125                 // 2000 / 16
#define NT        32                  // 512 / 16
#define KT_T      15                  // ceil(456/32) -> K padded to 480
#define KT_O      22                  // 704 / 32 exact

typedef __attribute__((ext_vector_type(16))) _Float16 v16h;
typedef __attribute__((ext_vector_type(8)))  float    v8f;

// ---------------------------------------------------------------------------
// Phase 0: perform_z = relu(exp_W @ perf + exp_b)
// ---------------------------------------------------------------------------
__global__ void k_perform_z(const float* __restrict__ W, const float* __restrict__ b,
                            const float* __restrict__ v, float* __restrict__ pz) {
  int t = threadIdx.x;            // 64 threads
  float s = b[t];
  for (int k = 0; k < ENCV; ++k) s += W[t * ENCV + k] * v[k];
  pz[t] = fmaxf(s, 0.f);
}

// ---------------------------------------------------------------------------
// Weight repack:
//  - Whh^T in f16, k-pair interleaved: elem (k,u) -> [(k>>1)*1024 + u*2 + (k&1)]
//    so the scan kernel reads one dword per 2 k-steps from LDS, conflict-free.
//  - recurrent column slices transposed (f32), summed biases.
// ---------------------------------------------------------------------------
__device__ __forceinline__ int whh_pack_idx(int k, int u) {
  return (k >> 1) * 1024 + u * 2 + (k & 1);
}

__global__ void k_pack_misc(const float* __restrict__ tWih, const float* __restrict__ oWih,
                            const float* __restrict__ tWhh, const float* __restrict__ oWhh,
                            const float* __restrict__ t_bih, const float* __restrict__ t_bhh,
                            const float* __restrict__ o_bih, const float* __restrict__ o_bhh,
                            _Float16* __restrict__ tWhhT, _Float16* __restrict__ oWhhT,
                            float* __restrict__ tRecT, float* __restrict__ oRecT,
                            float* __restrict__ biasT, float* __restrict__ biasO) {
  int idx = blockIdx.x * blockDim.x + threadIdx.x;
  if (idx < 128 * 512) {
    int k = idx >> 9, u = idx & 511;
    tWhhT[whh_pack_idx(k, u)] = (_Float16)tWhh[u * 128 + k];
    oWhhT[whh_pack_idx(k, u)] = (_Float16)oWhh[u * 128 + k];
  }
  if (idx < 11 * 512) {                        // recurrent column slices, transposed
    int j = idx >> 9, u = idx & 511;
    // beat LSTM: col 384 = prev_tempo, cols 393..402 = result_node
    tRecT[idx] = tWih[u * T_IN + (j == 0 ? 384 : 392 + j)];
    // out LSTM: cols 640..650 = prev_out
    oRecT[idx] = oWih[u * F_IN + 640 + j];
  }
  if (idx < 512) {
    biasT[idx] = t_bih[idx] + t_bhh[idx];
    biasO[idx] = o_bih[idx] + o_bhh[idx];
  }
}

// ---------------------------------------------------------------------------
// WMMA fragment index helpers (wave32 layouts per CDNA5 ISA 7.12.2).
// A 16x32 f16: lanes 0-15 row M=lane K 0..7/16..23; lanes 16-31 K 8..15/24..31
// B 32x16 f16: lane = N + 16*hi; same K order as A: K 0..7/16..23 (hi=0), 8..15/24..31 (hi=1)
// ---------------------------------------------------------------------------
__device__ __forceinline__ int a_klocal(int hi, int v, int pos) {
  return (v < 4) ? (8 * hi + 2 * v + pos) : (16 + 8 * hi + 2 * (v - 4) + pos);
}
__device__ __forceinline__ int b_klocal(int hi, int v, int pos) {
  return (v < 4) ? (8 * hi + 2 * v + pos) : (16 + 8 * hi + 2 * (v - 4) + pos);
}

// Pack A for beat-LSTM GEMM: X_t[i,k] over known features (be,me,ri,pz), K pad 480
__global__ void k_pack_At(const int* __restrict__ bnr, const int* __restrict__ mnr,
                          const float* __restrict__ be, const float* __restrict__ me,
                          const float* __restrict__ ri, const float* __restrict__ pz,
                          _Float16* __restrict__ Ap) {
  int idx = blockIdx.x * blockDim.x + threadIdx.x;
  const int total = MT * KT_T * 512;
  if (idx >= total) return;
  int tile = idx >> 9, rem = idx & 511, lane = rem >> 4, e = rem & 15;
  int mt = tile / KT_T, kt = tile - mt * KT_T;
  int i = mt * 16 + (lane & 15);
  int kl = a_klocal(lane >> 4, e >> 1, e & 1);
  int k = kt * 32 + kl;
  int b_ = bnr[i] - bnr[0], m_ = mnr[i] - mnr[0];
  float val = 0.f;
  if (k < 256)      val = be[b_ * BEAT_D + k];
  else if (k < 384) val = me[m_ * MEAS_D + (k - 256)];
  else if (k < 392) val = ri[b_ * RES_D + (k - 384)];
  else if (k < 456) val = pz[k - 392];
  Ap[idx] = (_Float16)val;
}

// Pack A for out-LSTM GEMM: X_o[i,k] over (ne,be,me,pz), K = 704
__global__ void k_pack_Ao(const int* __restrict__ bnr, const int* __restrict__ mnr,
                          const float* __restrict__ ne, const float* __restrict__ be,
                          const float* __restrict__ me, const float* __restrict__ pz,
                          _Float16* __restrict__ Ap) {
  int idx = blockIdx.x * blockDim.x + threadIdx.x;
  const int total = MT * KT_O * 512;
  if (idx >= total) return;
  int tile = idx >> 9, rem = idx & 511, lane = rem >> 4, e = rem & 15;
  int mt = tile / KT_O, kt = tile - mt * KT_O;
  int i = mt * 16 + (lane & 15);
  int kl = a_klocal(lane >> 4, e >> 1, e & 1);
  int k = kt * 32 + kl;
  int b_ = bnr[i] - bnr[0], m_ = mnr[i] - mnr[0];
  float val;
  if (k < 256)      val = ne[i * NOTE_D + k];
  else if (k < 512) val = be[b_ * BEAT_D + (k - 256)];
  else if (k < 640) val = me[m_ * MEAS_D + (k - 512)];
  else              val = pz[k - 640];
  Ap[idx] = (_Float16)val;
}

// Pack B = Wih^T restricted to known columns (skips recurrent cols)
__global__ void k_pack_Bt(const float* __restrict__ tWih, _Float16* __restrict__ Bp) {
  int idx = blockIdx.x * blockDim.x + threadIdx.x;
  const int total = KT_T * NT * 512;
  if (idx >= total) return;
  int tile = idx >> 9, rem = idx & 511, lane = rem >> 4, e = rem & 15;
  int kt = tile >> 5, nt = tile & 31;
  int k = kt * 32 + b_klocal(lane >> 4, e >> 1, e & 1);
  int u = nt * 16 + (lane & 15);
  float val = 0.f;
  if (k < 456) {
    int oc = (k < 384) ? k : ((k < 392) ? k + 1 : k + 11);  // skip cols 384, 393..402
    val = tWih[u * T_IN + oc];
  }
  Bp[idx] = (_Float16)val;
}

__global__ void k_pack_Bo(const float* __restrict__ oWih, _Float16* __restrict__ Bp) {
  int idx = blockIdx.x * blockDim.x + threadIdx.x;
  const int total = KT_O * NT * 512;
  if (idx >= total) return;
  int tile = idx >> 9, rem = idx & 511, lane = rem >> 4, e = rem & 15;
  int kt = tile >> 5, nt = tile & 31;
  int k = kt * 32 + b_klocal(lane >> 4, e >> 1, e & 1);
  int u = nt * 16 + (lane & 15);
  int oc = (k < 640) ? k : k + 11;                          // skip cols 640..650
  Bp[idx] = (_Float16)oWih[u * F_IN + oc];
}

// ---------------------------------------------------------------------------
// Phase 1 GEMM: Z[2000x512] = X @ Wih_known^T + (bih+bhh), one 16x16 C tile
// per wave via v_wmma_f32_16x16x32_f16.
// ---------------------------------------------------------------------------
__global__ void k_wmma_gemm(const _Float16* __restrict__ Ap, const _Float16* __restrict__ Bp,
                            const float* __restrict__ bias, float* __restrict__ Z, int KT) {
  int lane = threadIdx.x;                    // 32 threads = 1 wave
  int mt = blockIdx.x / NT, nt = blockIdx.x % NT;
  v8f acc = {};
  for (int kt = 0; kt < KT; ++kt) {
    v16h a = *(const v16h*)(Ap + (((size_t)(mt * KT + kt)) << 9) + (lane << 4));
    v16h b = *(const v16h*)(Bp + (((size_t)(kt * NT + nt)) << 9) + (lane << 4));
    acc = __builtin_amdgcn_wmma_f32_16x16x32_f16(false, a, false, b, (short)0, acc,
                                                 false, false);
  }
  int nl = lane & 15, hi = lane >> 4;
  int n = nt * 16 + nl;
  float bv = bias[n];
  #pragma unroll
  for (int r = 0; r < 8; ++r) {
    int m = mt * 16 + r + 8 * hi;            // D layout: VGPR r -> M=r (+8 for hi lanes)
    Z[(size_t)m * 512 + n] = acc[r] + bv;
  }
}

// ---------------------------------------------------------------------------
// Phase 2: sequential scan, one 512-thread block (16 waves).
//  - Whh^T (f16, pair-packed) lives in LDS: 256 KB of the WGP's 320 KB.
//  - Beat-LSTM + attention run ONLY on beat boundaries (uniform branch);
//    the reference discards their results otherwise, so this is exact.
// ---------------------------------------------------------------------------
__device__ __forceinline__ float sigm(float x) { return 1.f / (1.f + __expf(-x)); }

__global__ void __launch_bounds__(512)
k_seq(const float* __restrict__ Zt, const float* __restrict__ Zo,
      const _Float16* __restrict__ tWhhTg, const _Float16* __restrict__ oWhhTg,
      const float* __restrict__ tRecT, const float* __restrict__ oRecT,
      const float* __restrict__ tfcW, const float* __restrict__ tfcB,
      const float* __restrict__ attnW, const float* __restrict__ attnB,
      const float* __restrict__ attnCV,
      const float* __restrict__ fcW, const float* __restrict__ fcB,
      const int* __restrict__ bnr,
      float* __restrict__ out_buf, float* __restrict__ out_total) {
  __shared__ __align__(16) _Float16 tWhhL[128 * 512];   // 128 KB
  __shared__ __align__(16) _Float16 oWhhL[128 * 512];   // 128 KB
  __shared__ float th[128], tc[128], fh[128], fcl[128], nth[128], ntc[128];
  __shared__ float z[512], red[128];
  __shared__ float simb[N_NOTES];
  __shared__ float fcW_s[OUTF * 128], tfcW_s[128];
  __shared__ float attnW_s[OUTF * OUTF], attnB_s[OUTF], attnCV_s[OUTF], fcB_s[OUTF];
  __shared__ float prev_out[OUTF + 1], resN[OUTF];
  __shared__ float sPrevTempo, sNewTempo, sSimMax, sSimDen, sTfcB;
  __shared__ int   sPrevBeat, sPrevEnd;

  const int tid = threadIdx.x;
  const int bn0 = bnr[0];

  // stage Whh^T (f16 pair-packed) into LDS: 2 x 8192 dword4 copies
  {
    const uint4* ts = (const uint4*)tWhhTg; uint4* td = (uint4*)tWhhL;
    const uint4* os = (const uint4*)oWhhTg; uint4* od = (uint4*)oWhhL;
    for (int x = tid; x < 8192; x += 512) { td[x] = ts[x]; od[x] = os[x]; }
  }
  if (tid < 128) { th[tid] = tc[tid] = fh[tid] = fcl[tid] = 0.f; tfcW_s[tid] = tfcW[tid]; }
  for (int x = tid; x < OUTF * 128; x += 512) fcW_s[x] = fcW[x];
  if (tid < OUTF * OUTF) attnW_s[tid] = attnW[tid];
  if (tid < OUTF) { attnB_s[tid] = attnB[tid]; attnCV_s[tid] = attnCV[tid]; fcB_s[tid] = fcB[tid]; }
  if (tid < OUTF + 1) prev_out[tid] = 0.f;
  if (tid == 0) { sPrevTempo = 0.f; sPrevBeat = -1; sPrevEnd = 0; sTfcB = tfcB[0]; }
  __syncthreads();

  const unsigned* tWl = (const unsigned*)tWhhL;
  const unsigned* oWl = (const unsigned*)oWhhL;

  for (int i = 0; i < N_NOTES; ++i) {
    const int pe = sPrevEnd, pb = sPrevBeat;
    const float ptempo = sPrevTempo;
    const int cb = bnr[i] - bn0;
    const bool boundary = (cb > pb);          // uniform across the block

    // prefetch next step's precomputed gate rows (global_prefetch_b8)
    if (i + 1 < N_NOTES) {
      if (tid < 16)
        __builtin_prefetch((const char*)(Zo + (size_t)(i + 1) * 512) + tid * 128, 0, 0);
      else if (tid < 32)
        __builtin_prefetch((const char*)(Zt + (size_t)(i + 1) * 512) + (tid - 16) * 128, 0, 0);
    }

    if (boundary) {
      // ---- ContextAttention over ragged span [pe, i) ----
      if (tid == 0) {
        if (i > pe) {
          float mx = -3.4e38f;
          for (int t = pe; t < i; ++t) mx = fmaxf(mx, simb[t]);
          float den = 0.f;
          for (int t = pe; t < i; ++t) den += __expf(simb[t] - mx);
          sSimMax = mx; sSimDen = fmaxf(den, 1e-20f);
        } else sSimDen = 0.f;
      }
      __syncthreads();
      if (tid < OUTF) {
        float r = 0.f;
        if (sSimDen > 0.f) {
          float mx = sSimMax, inv = 1.f / sSimDen;
          for (int t = pe; t < i; ++t)
            r += __expf(simb[t] - mx) * inv * out_buf[t * OUTF + tid];
        }
        resN[tid] = r;
      }
      __syncthreads();

      // ---- beat-LSTM pre-activations: precomputed + recurrent cols + Whh@h ----
      {
        float acc = Zt[(size_t)i * 512 + tid];
        acc += tRecT[tid] * ptempo;
        #pragma unroll
        for (int j = 0; j < OUTF; ++j) acc += tRecT[(1 + j) * 512 + tid] * resN[j];
        #pragma unroll 8
        for (int k2 = 0; k2 < 64; ++k2) {
          union { unsigned u; _Float16 h[2]; } cv;
          cv.u = tWl[k2 * 512 + tid];
          acc += (float)cv.h[0] * th[2 * k2] + (float)cv.h[1] * th[2 * k2 + 1];
        }
        z[tid] = acc;
      }
      __syncthreads();
      if (tid < 128) {
        float ig = sigm(z[tid]), fg = sigm(z[128 + tid]);
        float g = tanhf(z[256 + tid]), og = sigm(z[384 + tid]);
        float c2 = fg * tc[tid] + ig * g;
        float h2 = og * tanhf(c2);
        nth[tid] = h2; ntc[tid] = c2;
        red[tid] = tfcW_s[tid] * h2;
      }
      __syncthreads();
      if (tid == 0) {
        float s = 0.f;
        for (int k = 0; k < 128; ++k) s += red[k];
        sNewTempo = s + sTfcB;
      }
      __syncthreads();
      if (tid < 128) { th[tid] = nth[tid]; tc[tid] = ntc[tid]; }
      if (tid == 0) { sPrevTempo = sNewTempo; sPrevEnd = i; sPrevBeat = cb; }
    }

    // ---- out-LSTM pre-activations (prev_out still holds last step's out_i) ----
    {
      float acc = Zo[(size_t)i * 512 + tid];
      #pragma unroll
      for (int j = 0; j < OUTF + 1; ++j) acc += oRecT[j * 512 + tid] * prev_out[j];
      #pragma unroll 8
      for (int k2 = 0; k2 < 64; ++k2) {
        union { unsigned u; _Float16 h[2]; } cv;
        cv.u = oWl[k2 * 512 + tid];
        acc += (float)cv.h[0] * fh[2 * k2] + (float)cv.h[1] * fh[2 * k2 + 1];
      }
      z[tid] = acc;
    }
    __syncthreads();
    if (tid < 128) {
      float ig = sigm(z[tid]), fg = sigm(z[128 + tid]);
      float g = tanhf(z[256 + tid]), og = sigm(z[384 + tid]);
      float c2 = fg * fcl[tid] + ig * g;
      float h2 = og * tanhf(c2);
      fh[tid] = h2; fcl[tid] = c2;     // unconditional commit
    }
    __syncthreads();
    if (tid < OUTF) {                   // out_fc = fc_W @ fh + fc_b
      float s = fcB_s[tid];
      for (int k = 0; k < 128; ++k) s += fcW_s[tid * 128 + k] * fh[k];
      red[tid] = s;
    }
    __syncthreads();
    if (tid == 0) {                     // sim_i = tanh(attn_W@out_fc+attn_b).attn_cv
      float si = 0.f;
      for (int r = 0; r < OUTF; ++r) {
        float a = attnB_s[r];
        for (int c = 0; c < OUTF; ++c) a += attnW_s[r * OUTF + c] * red[c];
        si += tanhf(a) * attnCV_s[r];
      }
      simb[i] = si;
    }
    const float ct = boundary ? sNewTempo : ptempo;     // committed tempo
    if (tid < OUTF) {
      out_buf[i * OUTF + tid] = red[tid];
      out_total[i * (OUTF + 1) + 1 + tid] = red[tid];
    }
    __syncthreads();                    // z_o reads of prev_out done; safe to overwrite
    if (tid < OUTF) prev_out[1 + tid] = red[tid];
    if (tid == 0) { prev_out[0] = ct; out_total[i * (OUTF + 1)] = ct; }
    __syncthreads();
  }
}

// ---------------------------------------------------------------------------
// Host launcher
// ---------------------------------------------------------------------------
extern "C" void kernel_launch(void* const* d_in, const int* in_sizes, int n_in,
                              void* d_out, int out_size, void* d_ws, size_t ws_size,
                              hipStream_t stream) {
  const float* ne   = (const float*)d_in[0];
  const float* be   = (const float*)d_in[1];
  const float* me   = (const float*)d_in[2];
  const float* ri   = (const float*)d_in[3];
  const float* perf = (const float*)d_in[4];
  const int*   bnr  = (const int*)d_in[5];
  const int*   mnr  = (const int*)d_in[6];
  const float* expW = (const float*)d_in[7];
  const float* expB = (const float*)d_in[8];
  const float* tWih = (const float*)d_in[9];
  const float* tWhh = (const float*)d_in[10];
  const float* tbih = (const float*)d_in[11];
  const float* tbhh = (const float*)d_in[12];
  const float* tfcW = (const float*)d_in[13];
  const float* tfcB = (const float*)d_in[14];
  const float* attnW = (const float*)d_in[15];
  const float* attnB = (const float*)d_in[16];
  const float* attnCV = (const float*)d_in[17];
  const float* oWih = (const float*)d_in[18];
  const float* oWhh = (const float*)d_in[19];
  const float* obih = (const float*)d_in[20];
  const float* obhh = (const float*)d_in[21];
  const float* fcW  = (const float*)d_in[22];
  const float* fcB  = (const float*)d_in[23];
  float* out = (float*)d_out;

  // workspace cursor allocator, 256 B aligned
  uintptr_t cur = (uintptr_t)d_ws;
  auto take = [&](size_t bytes) -> void* {
    cur = (cur + 255) & ~(uintptr_t)255;
    void* p = (void*)cur; cur += bytes; return p;
  };
  float*     pz     = (float*)take(ENC * 4);
  float*     Zt     = (float*)take((size_t)N_NOTES * 512 * 4);
  float*     Zo     = (float*)take((size_t)N_NOTES * 512 * 4);
  float*     outbuf = (float*)take((size_t)N_NOTES * OUTF * 4);
  _Float16*  At     = (_Float16*)take((size_t)MT * KT_T * 512 * 2);
  _Float16*  Ao     = (_Float16*)take((size_t)MT * KT_O * 512 * 2);
  _Float16*  Bt     = (_Float16*)take((size_t)KT_T * NT * 512 * 2);
  _Float16*  Bo     = (_Float16*)take((size_t)KT_O * NT * 512 * 2);
  _Float16*  tWhhT  = (_Float16*)take(128 * 512 * 2);
  _Float16*  oWhhT  = (_Float16*)take(128 * 512 * 2);
  float*     tRecT  = (float*)take(11 * 512 * 4);
  float*     oRecT  = (float*)take(11 * 512 * 4);
  float*     biasT  = (float*)take(512 * 4);
  float*     biasO  = (float*)take(512 * 4);

  // Phase 0: style expansion
  k_perform_z<<<1, 64, 0, stream>>>(expW, expB, perf, pz);

  // Weight repack (independent of pz)
  k_pack_misc<<<(128 * 512 + 255) / 256, 256, 0, stream>>>(
      tWih, oWih, tWhh, oWhh, tbih, tbhh, obih, obhh,
      tWhhT, oWhhT, tRecT, oRecT, biasT, biasO);
  k_pack_Bt<<<(KT_T * NT * 512 + 255) / 256, 256, 0, stream>>>(tWih, Bt);
  k_pack_Bo<<<(KT_O * NT * 512 + 255) / 256, 256, 0, stream>>>(oWih, Bo);

  // Feature packing (stream-ordered after pz)
  k_pack_At<<<(MT * KT_T * 512 + 255) / 256, 256, 0, stream>>>(bnr, mnr, be, me, ri, pz, At);
  k_pack_Ao<<<(MT * KT_O * 512 + 255) / 256, 256, 0, stream>>>(bnr, mnr, ne, be, me, pz, Ao);

  // Phase 1: batched pre-activation GEMMs on WMMA
  k_wmma_gemm<<<MT * NT, 32, 0, stream>>>(At, Bt, biasT, Zt, KT_T);
  k_wmma_gemm<<<MT * NT, 32, 0, stream>>>(Ao, Bo, biasO, Zo, KT_O);

  // Phase 2: sequential decoder scan
  k_seq<<<1, 512, 0, stream>>>(Zt, Zo, tWhhT, oWhhT, tRecT, oRecT,
                               tfcW, tfcB, attnW, attnB, attnCV, fcW, fcB,
                               bnr, outbuf, out);
  (void)in_sizes; (void)n_in; (void)out_size; (void)ws_size;
}
